// VoidAttention_1778116460615
// MI455X (gfx1250) — matrix operand order repaired and checked by reading, not compiled, not run on hardware
//
#include <hip/hip_runtime.h>
#include <hip/hip_bf16.h>

#define HEAD_DIM 64
#define HID      3072
#define NHEADS   48
#define NTOK     2048
#define TXT      256
#define VID      1792

typedef __attribute__((ext_vector_type(16))) __bf16 v16bf;
typedef __attribute__((ext_vector_type(8)))  __bf16 v8bf;
typedef __attribute__((ext_vector_type(8)))  float  v8f;

union FragBF { v16bf v; v8bf h[2]; };

static __device__ __forceinline__ __bf16 f2bf(float x) {
  union { float f; unsigned u; } a; a.f = x;
  unsigned r = a.u + 0x7FFFu + ((a.u >> 16) & 1u);
  union { unsigned short s; __bf16 b; } o; o.s = (unsigned short)(r >> 16);
  return o.b;
}

// A-fragment (16x32, MxK): lane L<16 -> row M=L, K = {0..7, 16..23}; lane L>=16 -> row M=L-16, K = {8..15, 24..31}
static __device__ __forceinline__ v16bf load_fragA(const __bf16* base, int ld, int row, int k0, int lane) {
  const int half = (lane >> 4) & 1;
  const __bf16* p = base + (size_t)row * (size_t)ld + k0 + half * 8;
  FragBF f;
  f.h[0] = *(const v8bf*)(p);
  f.h[1] = *(const v8bf*)(p + 16);
  return f.v;
}

// B-fragment (32x16, KxN): lane L<16 -> col N=L, K = {0..7, 16..23}; lane L>=16 -> col N=L-16, K = {8..15, 24..31}
// "nrow" in memory = column n of B (NT layout: memory holds [n][k] contiguous along k)
static __device__ __forceinline__ v16bf load_fragB(const __bf16* base, int ld, int nrow, int k0, int lane) {
  const int half = (lane >> 4) & 1;
  const __bf16* p = base + (size_t)nrow * (size_t)ld + k0 + half * 8;
  FragBF f;
  f.h[0] = *(const v8bf*)(p);
  f.h[1] = *(const v8bf*)(p + 16);
  return f.v;
}

static __device__ __forceinline__ v8f wmma_bf16(v16bf a, v16bf b, v8f c) {
  return __builtin_amdgcn_wmma_f32_16x16x32_bf16(false, a, false, b, (short)0, c, false, false);
}

// ---------------------------------------------------------------------------
// Stage 0: f32 -> bf16 conversion
// ---------------------------------------------------------------------------
__global__ void cvt_f32_bf16(const float* __restrict__ src, __bf16* __restrict__ dst, int n) {
  int i = (blockIdx.x * blockDim.x + threadIdx.x) * 4;
  if (i + 3 < n) {
    float4 v = *(const float4*)(src + i);
    dst[i + 0] = f2bf(v.x);
    dst[i + 1] = f2bf(v.y);
    dst[i + 2] = f2bf(v.z);
    dst[i + 3] = f2bf(v.w);
  } else {
    for (int j = i; j < n; ++j) dst[j] = f2bf(src[j]);
  }
}

// ---------------------------------------------------------------------------
// Stage 1: QKV GEMM (X @ W.T + b) with fused per-head layernorm + RoPE.
// Block tile 256(M) x 64(N); N-tile == one head. 8 waves, each wave:
// 32 rows x 64 cols = 2 M-subtiles x 4 N-subtiles (8 WMMA accumulators), so
// each B-fragment feeds two WMMAs (2x better compute density per load).
// blockIdx.z selects Q/K/V. Output bf16, head-major [h][token][64].
// ---------------------------------------------------------------------------
__global__ void __launch_bounds__(256)
qkv_gemm_ln_rope(const __bf16* __restrict__ Xb,
                 const __bf16* __restrict__ Wqb, const __bf16* __restrict__ Wkb,
                 const __bf16* __restrict__ Wvb,
                 const float* __restrict__ bq, const float* __restrict__ bk,
                 const float* __restrict__ bv,
                 const float* __restrict__ gq, const float* __restrict__ bgq,
                 const float* __restrict__ gk, const float* __restrict__ bgk,
                 const float* __restrict__ freqs,
                 __bf16* __restrict__ Qh, __bf16* __restrict__ Kh, __bf16* __restrict__ Vh) {
  const int which = blockIdx.z;
  const __bf16* W   = (which == 0) ? Wqb : ((which == 1) ? Wkb : Wvb);
  const float* bias = (which == 0) ? bq  : ((which == 1) ? bk  : bv);
  const float* g    = (which == 0) ? gq  : gk;
  const float* bg   = (which == 0) ? bgq : bgk;
  __bf16* dst       = (which == 0) ? Qh  : ((which == 1) ? Kh  : Vh);
  const bool doLN   = (which < 2);

  const int h  = blockIdx.y;
  const int n0 = h * HEAD_DIM;
  const int m0 = blockIdx.x * 256;
  const int lane = threadIdx.x & 31;
  const int wave = threadIdx.x >> 5;
  const int colbase = lane & 15;
  const int halfsel = (lane >> 4) & 1;
  const int rowbase = m0 + wave * 32;     // 32 rows per wave
  const int mrowA0 = rowbase + colbase;
  const int mrowA1 = rowbase + 16 + colbase;

  v8f acc[2][4] = {};
  for (int k0 = 0; k0 < HID; k0 += 32) {
    if (k0 + 32 < HID) {
      __builtin_prefetch((const void*)(Xb + (size_t)mrowA0 * HID + k0 + 32), 0, 3);
      __builtin_prefetch((const void*)(Xb + (size_t)mrowA1 * HID + k0 + 32), 0, 3);
    }
    v16bf a0 = load_fragA(Xb, HID, mrowA0, k0, lane);
    v16bf a1 = load_fragA(Xb, HID, mrowA1, k0, lane);
#pragma unroll
    for (int t = 0; t < 4; ++t) {
      v16bf b = load_fragB(W, HID, n0 + t * 16 + colbase, k0, lane);
      acc[0][t] = wmma_bf16(a0, b, acc[0][t]);
      acc[1][t] = wmma_bf16(a1, b, acc[1][t]);
    }
  }

#pragma unroll
  for (int ms = 0; ms < 2; ++ms) {
    // bias add (C-layout: reg r -> row r + 8*halfsel; col = t*16 + colbase)
#pragma unroll
    for (int t = 0; t < 4; ++t) {
      float bb = bias[n0 + t * 16 + colbase];
#pragma unroll
      for (int r = 0; r < 8; ++r) acc[ms][t][r] += bb;
    }

    if (doLN) {
#pragma unroll
      for (int r = 0; r < 8; ++r) {
        // layernorm over the 64-wide head: reduce across 16 lanes
        float s = acc[ms][0][r] + acc[ms][1][r] + acc[ms][2][r] + acc[ms][3][r];
        float q = acc[ms][0][r] * acc[ms][0][r] + acc[ms][1][r] * acc[ms][1][r] +
                  acc[ms][2][r] * acc[ms][2][r] + acc[ms][3][r] * acc[ms][3][r];
#pragma unroll
        for (int mk = 1; mk <= 8; mk <<= 1) {
          s += __shfl_xor(s, mk, 32);
          q += __shfl_xor(q, mk, 32);
        }
        float mu   = s * (1.0f / 64.0f);
        float var  = q * (1.0f / 64.0f) - mu * mu;
        float rstd = rsqrtf(var + 1e-5f);
#pragma unroll
        for (int t = 0; t < 4; ++t) {
          int col = t * 16 + colbase;
          acc[ms][t][r] = (acc[ms][t][r] - mu) * rstd * g[col] + bg[col];
        }
        // RoPE (reference quirk: freqs indexed by head, not position) on vid rows
        int row = rowbase + ms * 16 + r + 8 * halfsel;
        if (row >= TXT) {
#pragma unroll
          for (int t = 0; t < 4; ++t) {
            int col = t * 16 + colbase;
            int jj = col >> 1;
            int aa = col & 1;
            float mine = acc[ms][t][r];
            float part = __shfl_xor(mine, 1, 32);
            float xe = aa ? part : mine;
            float xo = aa ? mine : part;
            const float* F = freqs + (((size_t)(h * 32 + jj) * 2 + aa) * 2);
            acc[ms][t][r] = F[0] * xe + F[1] * xo;
          }
        }
      }
    }

    // store bf16 head-major [h][token][64]
#pragma unroll
    for (int r = 0; r < 8; ++r) {
      int row = rowbase + ms * 16 + r + 8 * halfsel;
#pragma unroll
      for (int t = 0; t < 4; ++t) {
        int col = t * 16 + colbase;
        dst[((size_t)h * NTOK + row) * HEAD_DIM + col] = f2bf(acc[ms][t][r]);
      }
    }
  }
}

// ---------------------------------------------------------------------------
// Stage 2: flash attention per head. 4 waves x 16 query rows = 64 rows/block.
// Key chunks of 64. S via WMMA from global K; P staged through per-wave LDS;
// V transposed into LDS so PV B-fragments are contiguous.
// ---------------------------------------------------------------------------
__global__ void __launch_bounds__(128)
flash_attn(const __bf16* __restrict__ Qg, const __bf16* __restrict__ Kg,
           const __bf16* __restrict__ Vg, __bf16* __restrict__ Ab) {
  __shared__ __align__(16) __bf16 Vt[HEAD_DIM][72];  // [d][key], padded stride
  __shared__ __align__(16) __bf16 Pl[4][16][72];     // per-wave P tile [qrow][key]

  const int h  = blockIdx.y;
  const int m0 = blockIdx.x * 64;
  const int tid  = threadIdx.x;
  const int lane = tid & 31;
  const int wave = tid >> 5;
  const int colbase = lane & 15;
  const int halfsel = (lane >> 4) & 1;

  const __bf16* Q = Qg + (size_t)h * NTOK * HEAD_DIM;
  const __bf16* K = Kg + (size_t)h * NTOK * HEAD_DIM;
  const __bf16* V = Vg + (size_t)h * NTOK * HEAD_DIM;

  const int qrow = m0 + wave * 16 + colbase;
  v16bf qa0 = load_fragA(Q, HEAD_DIM, qrow, 0, lane);
  v16bf qa1 = load_fragA(Q, HEAD_DIM, qrow, 32, lane);

  float mrun[8], lrun[8];
#pragma unroll
  for (int r = 0; r < 8; ++r) { mrun[r] = -1e30f; lrun[r] = 0.0f; }
  v8f o[4] = {};

  const float scale = 0.125f;  // 1/sqrt(64)

  for (int n0 = 0; n0 < NTOK; n0 += 64) {
    __syncthreads();  // previous iteration done reading Vt
    {   // cooperative transpose load of V[n0..n0+63][0..63] -> Vt[d][key]
      int vr = tid >> 3;        // 0..15
      int vc = (tid & 7) * 8;   // 0,8,...,56
#pragma unroll
      for (int i = 0; i < 4; ++i) {
        int key = vr + i * 16;
        v8bf vv = *(const v8bf*)(V + (size_t)(n0 + key) * HEAD_DIM + vc);
#pragma unroll
        for (int e = 0; e < 8; ++e) Vt[vc + e][key] = vv[e];
      }
    }
    __syncthreads();

    // S = scale * Q K^T  (4 subtiles of 16x16 over 64 keys)
    v8f st[4] = {};
#pragma unroll
    for (int t = 0; t < 4; ++t) {
      v16bf b0 = load_fragB(K, HEAD_DIM, n0 + t * 16 + colbase, 0, lane);
      st[t] = wmma_bf16(qa0, b0, st[t]);
      v16bf b1 = load_fragB(K, HEAD_DIM, n0 + t * 16 + colbase, 32, lane);
      st[t] = wmma_bf16(qa1, b1, st[t]);
    }

    // online softmax per row (reg r + lane half selects the row)
#pragma unroll
    for (int r = 0; r < 8; ++r) {
      float mx = -1e30f;
#pragma unroll
      for (int t = 0; t < 4; ++t) { st[t][r] *= scale; mx = fmaxf(mx, st[t][r]); }
#pragma unroll
      for (int mk = 1; mk <= 8; mk <<= 1) mx = fmaxf(mx, __shfl_xor(mx, mk, 32));
      float mnew = fmaxf(mrun[r], mx);
      float corr = __expf(mrun[r] - mnew);
      mrun[r] = mnew;
      float rs = 0.0f;
#pragma unroll
      for (int t = 0; t < 4; ++t) { st[t][r] = __expf(st[t][r] - mnew); rs += st[t][r]; }
#pragma unroll
      for (int mk = 1; mk <= 8; mk <<= 1) rs += __shfl_xor(rs, mk, 32);
      lrun[r] = lrun[r] * corr + rs;
#pragma unroll
      for (int t = 0; t < 4; ++t) o[t][r] *= corr;
      // stash P (bf16) for A-fragment reload
#pragma unroll
      for (int t = 0; t < 4; ++t)
        Pl[wave][r + 8 * halfsel][t * 16 + colbase] = f2bf(st[t][r]);
    }

    asm volatile("s_wait_dscnt 0" ::: "memory");  // cross-lane LDS RAW within wave

    // O += P @ V  (A from Pl, B from Vt)
    v16bf pa0 = load_fragA(&Pl[wave][0][0], 72, colbase, 0, lane);
    v16bf pa1 = load_fragA(&Pl[wave][0][0], 72, colbase, 32, lane);
#pragma unroll
    for (int td = 0; td < 4; ++td) {
      v16bf vb0 = load_fragB(&Vt[0][0], 72, td * 16 + colbase, 0, lane);
      o[td] = wmma_bf16(pa0, vb0, o[td]);
      v16bf vb1 = load_fragB(&Vt[0][0], 72, td * 16 + colbase, 32, lane);
      o[td] = wmma_bf16(pa1, vb1, o[td]);
    }
  }

  // normalize and store to token-major bf16 attn buffer [token][HID]
#pragma unroll
  for (int r = 0; r < 8; ++r) {
    float inv = 1.0f / lrun[r];
    int row = m0 + wave * 16 + r + 8 * halfsel;
#pragma unroll
    for (int td = 0; td < 4; ++td) {
      int col = h * HEAD_DIM + td * 16 + colbase;
      Ab[(size_t)row * HID + col] = f2bf(o[td][r] * inv);
    }
  }
}

// ---------------------------------------------------------------------------
// Stage 3: output projection attn @ Wo.T + bo, with vid-first / txt-last
// reorder. Same 2x M register blocking as stage 1.
// ---------------------------------------------------------------------------
__global__ void __launch_bounds__(256)
out_proj(const __bf16* __restrict__ Ab, const __bf16* __restrict__ Wob,
         const float* __restrict__ bo, float* __restrict__ out) {
  const int n0 = blockIdx.y * 64;
  const int m0 = blockIdx.x * 256;
  const int lane = threadIdx.x & 31;
  const int wave = threadIdx.x >> 5;
  const int colbase = lane & 15;
  const int halfsel = (lane >> 4) & 1;
  const int rowbase = m0 + wave * 32;
  const int mrowA0 = rowbase + colbase;
  const int mrowA1 = rowbase + 16 + colbase;

  v8f acc[2][4] = {};
  for (int k0 = 0; k0 < HID; k0 += 32) {
    if (k0 + 32 < HID) {
      __builtin_prefetch((const void*)(Ab + (size_t)mrowA0 * HID + k0 + 32), 0, 3);
      __builtin_prefetch((const void*)(Ab + (size_t)mrowA1 * HID + k0 + 32), 0, 3);
    }
    v16bf a0 = load_fragA(Ab, HID, mrowA0, k0, lane);
    v16bf a1 = load_fragA(Ab, HID, mrowA1, k0, lane);
#pragma unroll
    for (int t = 0; t < 4; ++t) {
      v16bf b = load_fragB(Wob, HID, n0 + t * 16 + colbase, k0, lane);
      acc[0][t] = wmma_bf16(a0, b, acc[0][t]);
      acc[1][t] = wmma_bf16(a1, b, acc[1][t]);
    }
  }

#pragma unroll
  for (int ms = 0; ms < 2; ++ms) {
#pragma unroll
    for (int r = 0; r < 8; ++r) {
      int row = rowbase + ms * 16 + r + 8 * halfsel;
      int outrow = (row >= TXT) ? (row - TXT) : (row + VID);
#pragma unroll
      for (int t = 0; t < 4; ++t) {
        int col = n0 + t * 16 + colbase;
        out[(size_t)outrow * HID + col] = acc[ms][t][r] + bo[col];
      }
    }
  }
}

// ---------------------------------------------------------------------------
extern "C" void kernel_launch(void* const* d_in, const int* in_sizes, int n_in,
                              void* d_out, int out_size, void* d_ws, size_t ws_size,
                              hipStream_t stream) {
  (void)in_sizes; (void)n_in; (void)out_size; (void)ws_size;

  const float* X     = (const float*)d_in[0];
  const float* freqs = (const float*)d_in[1];
  const float* Wq    = (const float*)d_in[2];
  const float* bq    = (const float*)d_in[3];
  const float* Wk    = (const float*)d_in[4];
  const float* bk    = (const float*)d_in[5];
  const float* Wv    = (const float*)d_in[6];
  const float* bv    = (const float*)d_in[7];
  const float* Wo    = (const float*)d_in[8];
  const float* bo    = (const float*)d_in[9];
  const float* gq    = (const float*)d_in[10];
  const float* bgq   = (const float*)d_in[11];
  const float* gk    = (const float*)d_in[12];
  const float* bgk   = (const float*)d_in[13];
  float* out = (float*)d_out;

  char* ws = (char*)d_ws;
  size_t off = 0;
  auto wsalloc = [&](size_t bytes) -> char* {
    char* p = ws + off;
    off += (bytes + 255) & ~(size_t)255;
    return p;
  };
  const size_t actBytes = (size_t)NTOK * HID * sizeof(__bf16);  // 12.58 MB
  const size_t wBytes   = (size_t)HID * HID * sizeof(__bf16);   // 18.87 MB
  __bf16* Xb  = (__bf16*)wsalloc(actBytes);
  __bf16* Wqb = (__bf16*)wsalloc(wBytes);
  __bf16* Wkb = (__bf16*)wsalloc(wBytes);
  __bf16* Wvb = (__bf16*)wsalloc(wBytes);
  __bf16* Wob = (__bf16*)wsalloc(wBytes);
  __bf16* Qh  = (__bf16*)wsalloc(actBytes);
  __bf16* Kh  = (__bf16*)wsalloc(actBytes);
  __bf16* Vh  = (__bf16*)wsalloc(actBytes);
  __bf16* Ab  = (__bf16*)wsalloc(actBytes);

  // Stage 0: conversions
  {
    int nX = NTOK * HID;
    int nW = HID * HID;
    int thr = 256;
    cvt_f32_bf16<<<(nX / 4 + thr - 1) / thr, thr, 0, stream>>>(X,  Xb,  nX);
    cvt_f32_bf16<<<(nW / 4 + thr - 1) / thr, thr, 0, stream>>>(Wq, Wqb, nW);
    cvt_f32_bf16<<<(nW / 4 + thr - 1) / thr, thr, 0, stream>>>(Wk, Wkb, nW);
    cvt_f32_bf16<<<(nW / 4 + thr - 1) / thr, thr, 0, stream>>>(Wv, Wvb, nW);
    cvt_f32_bf16<<<(nW / 4 + thr - 1) / thr, thr, 0, stream>>>(Wo, Wob, nW);
  }

  // Stage 1: QKV projection + LN + RoPE (z: 0=Q, 1=K, 2=V)
  {
    dim3 grid(NTOK / 256, NHEADS, 3);
    qkv_gemm_ln_rope<<<grid, 256, 0, stream>>>(Xb, Wqb, Wkb, Wvb,
                                               bq, bk, bv,
                                               gq, bgq, gk, bgk,
                                               freqs, Qh, Kh, Vh);
  }

  // Stage 2: attention
  {
    dim3 grid(NTOK / 64, NHEADS);
    flash_attn<<<grid, 128, 0, stream>>>(Qh, Kh, Vh, Ab);
  }

  // Stage 3: output projection + reorder
  {
    dim3 grid(NTOK / 256, HID / 64);
    out_proj<<<grid, 256, 0, stream>>>(Ab, Wob, bo, out);
  }
}
